// Alignn_56667798504232
// MI455X (gfx1250) — compile-verified, matrix operand order checked by reading, not run on hardware
//
#include <hip/hip_runtime.h>
#include <hip/hip_bf16.h>

typedef __attribute__((ext_vector_type(16))) _Float16 v16h;
typedef __attribute__((ext_vector_type(8)))  float    v8f;
typedef __attribute__((ext_vector_type(4)))  unsigned v4u;
typedef __attribute__((ext_vector_type(8)))  unsigned v8u;

#define NNODES   50000
#define NEDGES   1200000
#define HDIM     64
#define NGRAPHS  64
#define NCLASSES 10
#define PI_F     3.14159265358979f
#define LOG2E_F  1.44269504088896f
#define LN2_F    0.69314718055995f
#define BSWZ_WORDS 2048   // 2(kb) * 4(nt) * 2(h) * 16(l) * 8(g) packed dwords

// ---------------------------------------------------------------- utilities

__device__ __forceinline__ void wave_lds_fence() {
  // intra-wave LDS producer->consumer ordering (CDNA5 split DS counter)
  __builtin_amdgcn_wave_barrier();
  asm volatile("s_wait_dscnt 0" ::: "memory");
  __builtin_amdgcn_wave_barrier();
}

__device__ __forceinline__ int kmap(int g, int h) {
  // 16-bit A-matrix 16x32 VGPR layout (ISA 7.12.2)
  return (g < 4) ? (8 * h + 2 * g) : (16 + 8 * h + 2 * (g - 4));
}

// A fragment: two contiguous 16B LDS chunks per lane -> 2x ds_load_b128
__device__ __forceinline__ v16h load_a_vec(const _Float16* tile, int l, int h, int kb) {
  const _Float16* base = tile + l * HDIM + 32 * kb + 8 * h;
  v4u lo = *reinterpret_cast<const v4u*>(base);
  v4u hi = *reinterpret_cast<const v4u*>(base + 16);
  v8u w  = __builtin_shufflevector(lo, hi, 0, 1, 2, 3, 4, 5, 6, 7);
  return __builtin_bit_cast(v16h, w);
}

// stage a 64x64 f32 weight matrix into LDS, pre-swizzled to B-fragment order:
// dword i = {f16(W[k][n]), f16(W[k+1][n])}, i = (((kb*4+nt)*2+h)*16 + l)*8 + g
__device__ __forceinline__ void stage_b_swz(const float* __restrict__ W,
                                            unsigned* sBw, int tid) {
  for (int i = tid; i < BSWZ_WORDS; i += 256) {
    int g  = i & 7;
    int l  = (i >> 3) & 15;
    int h  = (i >> 7) & 1;
    int nt = (i >> 8) & 3;
    int kb = (i >> 10) & 1;
    int k  = 32 * kb + kmap(g, h);
    int n  = 16 * nt + l;
    _Float16 lo = (_Float16)W[k * HDIM + n];
    _Float16 hi = (_Float16)W[(k + 1) * HDIM + n];
    unsigned u = (unsigned)(*(unsigned short*)&lo) |
                 ((unsigned)(*(unsigned short*)&hi) << 16);
    sBw[i] = u;
  }
}

// B fragment: 8 consecutive dwords (32B aligned) -> 2x ds_load_b128, zero repacking
__device__ __forceinline__ v16h load_b_swz(const unsigned* sBw, int l, int h, int kb, int nt) {
  const unsigned* p = sBw + ((((kb * 4 + nt) * 2 + h) * 16) + l) * 8;
  v4u lo = *reinterpret_cast<const v4u*>(p);
  v4u hi = *reinterpret_cast<const v4u*>(p + 4);
  v8u w  = __builtin_shufflevector(lo, hi, 0, 1, 2, 3, 4, 5, 6, 7);
  return __builtin_bit_cast(v16h, w);
}

__device__ __forceinline__ v8f wmma16(v16h a, v16h b, v8f c) {
  return __builtin_amdgcn_wmma_f32_16x16x32_f16(false, a, false, b, (short)0, c, false, false);
}

__device__ __forceinline__ float softplus_f(float x) {
  // branch-free stable softplus using raw base-2 HW transcendentals:
  // max(x,0) + ln2*log2(1 + exp2(-log2e*|x|)); argument of log2 is in (1,2],
  // so no denorm/inf guards are needed (|x| and negate are free modifiers).
  float t = __builtin_amdgcn_exp2f(-LOG2E_F * fabsf(x));
  return fmaxf(x, 0.0f) + LN2_F * __builtin_amdgcn_logf(1.0f + t);
}

// bf tile: each lane owns half a row (row = lane>>1, 32 cols)
__device__ __forceinline__ void compute_bf_tile(const float* __restrict__ d, int eBase,
                                                int lane, _Float16* tile) {
  int row = lane >> 1;
  int c0  = (lane & 1) * 32;
  float dd  = d[eBase + row];
  float dc  = fminf(fmaxf(dd, 0.0f), 1.0f);
  float cut = 0.5f * (__cosf(PI_F * dc) + 1.0f);
#pragma unroll
  for (int j = 0; j < 32; ++j) {
    int col = c0 + j;
    float t = dd - (float)col * (1.0f / 63.0f);
    // exp(-64 t^2) via exp2
    tile[row * HDIM + col] = (_Float16)(__builtin_amdgcn_exp2f(-64.0f * LOG2E_F * t * t) * cut);
  }
}

// ---------------------------------------------------------------- kernels

__global__ void zero_kernel(float* p, int n) {
  int i = blockIdx.x * blockDim.x + threadIdx.x;
  if (i < n) p[i] = 0.0f;
}

__global__ void f32_to_f16_kernel(const float* __restrict__ in, _Float16* __restrict__ out, int n) {
  int i = blockIdx.x * blockDim.x + threadIdx.x;
  if (i < n) out[i] = (_Float16)in[i];
}

// x0 = scatter_add( mlp(bf) ) : bf -> W1 -> softplus -> W2 -> atomicAdd x[dst]
__global__ void __launch_bounds__(256) embed_kernel(
    const float* __restrict__ d,
    const float* __restrict__ W1, const float* __restrict__ B1,
    const float* __restrict__ W2, const float* __restrict__ B2,
    const int* __restrict__ dst, float* __restrict__ x) {
  __shared__ __align__(32) unsigned sW1[BSWZ_WORDS];
  __shared__ __align__(32) unsigned sW2[BSWZ_WORDS];
  __shared__ float sB1[HDIM], sB2[HDIM];
  __shared__ __align__(16) _Float16 t0[8][16 * HDIM];
  __shared__ __align__(16) _Float16 t1[8][16 * HDIM];

  int tid = threadIdx.x;
  stage_b_swz(W1, sW1, tid);
  stage_b_swz(W2, sW2, tid);
  if (tid < HDIM) { sB1[tid] = B1[tid]; sB2[tid] = B2[tid]; }
  __syncthreads();

  int wave = tid >> 5, lane = tid & 31;
  int l = lane & 15, h = lane >> 4;
  int eBase = (blockIdx.x * 8 + wave) * 16;
  if (eBase >= NEDGES) return;

  _Float16* bfT = t0[wave];
  _Float16* h1T = t1[wave];

  compute_bf_tile(d, eBase, lane, bfT);
  wave_lds_fence();

  // h1 = softplus(bf @ W1 + B1); A fragments hoisted across N-tiles
  {
    v16h a0 = load_a_vec(bfT, l, h, 0);
    v16h a1 = load_a_vec(bfT, l, h, 1);
#pragma unroll
    for (int nt = 0; nt < 4; ++nt) {
      v8f acc = {};
      acc = wmma16(a0, load_b_swz(sW1, l, h, 0, nt), acc);
      acc = wmma16(a1, load_b_swz(sW1, l, h, 1, nt), acc);
      int col = 16 * nt + l;
      float bias = sB1[col];
#pragma unroll
      for (int r = 0; r < 8; ++r)
        h1T[(r + 8 * h) * HDIM + col] = (_Float16)softplus_f(acc[r] + bias);
    }
  }
  wave_lds_fence();

  // out = h1 @ W2 + B2 -> scatter to x[dst]
  {
    v16h a0 = load_a_vec(h1T, l, h, 0);
    v16h a1 = load_a_vec(h1T, l, h, 1);
#pragma unroll
    for (int nt = 0; nt < 4; ++nt) {
      v8f acc = {};
      acc = wmma16(a0, load_b_swz(sW2, l, h, 0, nt), acc);
      acc = wmma16(a1, load_b_swz(sW2, l, h, 1, nt), acc);
      int col = 16 * nt + l;
      float bias = sB2[col];
#pragma unroll
      for (int r = 0; r < 8; ++r) {
        int m = r + 8 * h;
        atomicAdd(&x[dst[eBase + m] * HDIM + col], acc[r] + bias);
      }
    }
  }
}

// one conv layer, edge side: filt = mlp(bf); m = (x[src]@Win + bin)*filt; agg[dst] += m
__global__ void __launch_bounds__(256) conv_edge_kernel(
    const float* __restrict__ d,
    const float* __restrict__ fW1, const float* __restrict__ fB1,
    const float* __restrict__ fW2, const float* __restrict__ fB2,
    const float* __restrict__ Win, const float* __restrict__ Bin,
    const int* __restrict__ src, const int* __restrict__ dst,
    const _Float16* __restrict__ xin, float* __restrict__ agg) {
  __shared__ __align__(32) unsigned sW1[BSWZ_WORDS];
  __shared__ __align__(32) unsigned sW2[BSWZ_WORDS];
  __shared__ __align__(32) unsigned sWi[BSWZ_WORDS];
  __shared__ float sB1[HDIM], sB2[HDIM], sBi[HDIM];
  __shared__ __align__(16) _Float16 t0[8][16 * HDIM];  // bf, then gathered x[src]
  __shared__ __align__(16) _Float16 t1[8][16 * HDIM];  // softplus hidden

  int tid = threadIdx.x;
  stage_b_swz(fW1, sW1, tid);
  stage_b_swz(fW2, sW2, tid);
  stage_b_swz(Win, sWi, tid);
  if (tid < HDIM) { sB1[tid] = fB1[tid]; sB2[tid] = fB2[tid]; sBi[tid] = Bin[tid]; }
  __syncthreads();

  int wave = tid >> 5, lane = tid & 31;
  int l = lane & 15, h = lane >> 4;
  int eBase = (blockIdx.x * 8 + wave) * 16;
  if (eBase >= NEDGES) return;

  _Float16* aT = t0[wave];
  _Float16* hT = t1[wave];

  // ---- radial filter: filt = softplus(bf@fW1+fb1)@fW2 + fb2 (kept in registers)
  compute_bf_tile(d, eBase, lane, aT);
  wave_lds_fence();
  {
    v16h a0 = load_a_vec(aT, l, h, 0);
    v16h a1 = load_a_vec(aT, l, h, 1);
#pragma unroll
    for (int nt = 0; nt < 4; ++nt) {
      v8f acc = {};
      acc = wmma16(a0, load_b_swz(sW1, l, h, 0, nt), acc);
      acc = wmma16(a1, load_b_swz(sW1, l, h, 1, nt), acc);
      int col = 16 * nt + l;
      float bias = sB1[col];
#pragma unroll
      for (int r = 0; r < 8; ++r)
        hT[(r + 8 * h) * HDIM + col] = (_Float16)softplus_f(acc[r] + bias);
    }
  }
  wave_lds_fence();

  v8f filt[4];
  {
    v16h a0 = load_a_vec(hT, l, h, 0);
    v16h a1 = load_a_vec(hT, l, h, 1);
#pragma unroll
    for (int nt = 0; nt < 4; ++nt) {
      v8f acc = {};
      acc = wmma16(a0, load_b_swz(sW2, l, h, 0, nt), acc);
      acc = wmma16(a1, load_b_swz(sW2, l, h, 1, nt), acc);
      float bias = sB2[16 * nt + l];
#pragma unroll
      for (int r = 0; r < 8; ++r) acc[r] += bias;
      filt[nt] = acc;
    }
  }
  wave_lds_fence();  // bf-tile reads retired before t0 is overwritten below

  // ---- gather x[src] (f16, 128B/row) into t0 via 16B chunks
#pragma unroll
  for (int i = lane; i < 16 * 8; i += 32) {
    int row = i >> 3, ch = i & 7;
    int s = src[eBase + row];
    *reinterpret_cast<uint4*>(&aT[row * HDIM + ch * 8]) =
        *reinterpret_cast<const uint4*>(xin + (size_t)s * HDIM + ch * 8);
  }
  wave_lds_fence();

  // ---- m = (xs @ Win + bin) * filt -> atomic scatter into agg[dst]
  {
    v16h a0 = load_a_vec(aT, l, h, 0);
    v16h a1 = load_a_vec(aT, l, h, 1);
#pragma unroll
    for (int nt = 0; nt < 4; ++nt) {
      v8f acc = {};
      acc = wmma16(a0, load_b_swz(sWi, l, h, 0, nt), acc);
      acc = wmma16(a1, load_b_swz(sWi, l, h, 1, nt), acc);
      int col = 16 * nt + l;
      float bias = sBi[col];
#pragma unroll
      for (int r = 0; r < 8; ++r) {
        int m = r + 8 * h;
        float mv = (acc[r] + bias) * filt[nt][r];
        atomicAdd(&agg[dst[eBase + m] * HDIM + col], mv);
      }
    }
  }
}

// node side: x = relu(agg @ Wout + bout), written as f16 (next gather) and f32 (pooling)
__global__ void __launch_bounds__(256) conv_node_kernel(
    const float* __restrict__ agg,
    const float* __restrict__ Wout, const float* __restrict__ Bout,
    _Float16* __restrict__ xf16, float* __restrict__ xf32) {
  __shared__ __align__(32) unsigned sW[BSWZ_WORDS];
  __shared__ float sB[HDIM];
  __shared__ __align__(16) _Float16 t0[8][16 * HDIM];

  int tid = threadIdx.x;
  stage_b_swz(Wout, sW, tid);
  if (tid < HDIM) sB[tid] = Bout[tid];
  __syncthreads();

  int wave = tid >> 5, lane = tid & 31;
  int l = lane & 15, h = lane >> 4;
  int nBase = (blockIdx.x * 8 + wave) * 16;
  if (nBase >= NNODES) return;  // 50000 % 16 == 0, full tiles only

  _Float16* aT = t0[wave];
#pragma unroll
  for (int i = lane; i < 16 * HDIM; i += 32)
    aT[i] = (_Float16)agg[(size_t)nBase * HDIM + i];
  wave_lds_fence();

  {
    v16h a0 = load_a_vec(aT, l, h, 0);
    v16h a1 = load_a_vec(aT, l, h, 1);
#pragma unroll
    for (int nt = 0; nt < 4; ++nt) {
      v8f acc = {};
      acc = wmma16(a0, load_b_swz(sW, l, h, 0, nt), acc);
      acc = wmma16(a1, load_b_swz(sW, l, h, 1, nt), acc);
      int col = 16 * nt + l;
      float bias = sB[col];
#pragma unroll
      for (int r = 0; r < 8; ++r) {
        int m = r + 8 * h;
        float v = fmaxf(acc[r] + bias, 0.0f);
        size_t o = (size_t)(nBase + m) * HDIM + col;
        xf16[o] = (_Float16)v;
        xf32[o] = v;
      }
    }
  }
}

__global__ void pool_kernel(const float* __restrict__ x, const int* __restrict__ gid,
                            float* __restrict__ pooled, float* __restrict__ counts) {
  int idx = blockIdx.x * blockDim.x + threadIdx.x;
  if (idx >= NNODES * HDIM) return;
  int node = idx >> 6, col = idx & 63;
  int g = gid[node];
  atomicAdd(&pooled[g * HDIM + col], x[idx]);
  if (col == 0) atomicAdd(&counts[g], 1.0f);
}

__global__ void head_kernel(const float* __restrict__ pooled, const float* __restrict__ counts,
                            const float* __restrict__ Wfc, const float* __restrict__ bfc,
                            float* __restrict__ out) {
  int g = threadIdx.x;
  if (g >= NGRAPHS) return;
  float inv = 1.0f / fmaxf(counts[g], 1.0f);
  float logit[NCLASSES];
#pragma unroll
  for (int c = 0; c < NCLASSES; ++c) logit[c] = bfc[c];
  for (int k = 0; k < HDIM; ++k) {
    float v = pooled[g * HDIM + k] * inv;
#pragma unroll
    for (int c = 0; c < NCLASSES; ++c) logit[c] += v * Wfc[k * NCLASSES + c];
  }
  float mx = logit[0];
#pragma unroll
  for (int c = 1; c < NCLASSES; ++c) mx = fmaxf(mx, logit[c]);
  float s = 0.0f;
#pragma unroll
  for (int c = 0; c < NCLASSES; ++c) s += __expf(logit[c] - mx);
  float lse = mx + __logf(s);
#pragma unroll
  for (int c = 0; c < NCLASSES; ++c) out[g * NCLASSES + c] = logit[c] - lse;
}

// ---------------------------------------------------------------- launch

extern "C" void kernel_launch(void* const* d_in, const int* in_sizes, int n_in,
                              void* d_out, int out_size, void* d_ws, size_t ws_size,
                              hipStream_t stream) {
  (void)in_sizes; (void)n_in; (void)out_size; (void)ws_size;
  const float* d     = (const float*)d_in[0];
  const float* Wemb1 = (const float*)d_in[1];
  const float* bemb1 = (const float*)d_in[2];
  const float* Wemb2 = (const float*)d_in[3];
  const float* bemb2 = (const float*)d_in[4];
  const float* fW1   = (const float*)d_in[5];
  const float* fB1   = (const float*)d_in[6];
  const float* fW2   = (const float*)d_in[7];
  const float* fB2   = (const float*)d_in[8];
  const float* Win   = (const float*)d_in[9];
  const float* Bin   = (const float*)d_in[10];
  const float* Wout  = (const float*)d_in[11];
  const float* Bout  = (const float*)d_in[12];
  const float* Wfc   = (const float*)d_in[13];
  const float* bfc   = (const float*)d_in[14];
  const int*   src   = (const int*)d_in[15];
  const int*   dst   = (const int*)d_in[16];
  const int*   gid   = (const int*)d_in[17];

  char* ws = (char*)d_ws;
  size_t off = 0;
  auto carve = [&](size_t bytes) -> void* {
    void* p = ws + off;
    off = (off + bytes + 255) & ~(size_t)255;
    return p;
  };
  float*    xf32   = (float*)carve((size_t)NNODES * HDIM * sizeof(float));
  float*    agg    = (float*)carve((size_t)NNODES * HDIM * sizeof(float));
  _Float16* xf16   = (_Float16*)carve((size_t)NNODES * HDIM * sizeof(_Float16));
  float*    pooled = (float*)carve((size_t)NGRAPHS * HDIM * sizeof(float));
  float*    counts = (float*)carve((size_t)NGRAPHS * sizeof(float));

  const int nfeat = NNODES * HDIM;
  const int featBlocks = (nfeat + 255) / 256;
  const int edgeBlocks = (NEDGES / 16 + 7) / 8;   // 9375, 8 waves x 16-edge tiles
  const int nodeBlocks = (NNODES / 16 + 7) / 8;   // 391

  zero_kernel<<<featBlocks, 256, 0, stream>>>(xf32, nfeat);
  zero_kernel<<<(NGRAPHS * HDIM + 255) / 256, 256, 0, stream>>>(pooled, NGRAPHS * HDIM);
  zero_kernel<<<1, 64, 0, stream>>>(counts, NGRAPHS);

  embed_kernel<<<edgeBlocks, 256, 0, stream>>>(d, Wemb1, bemb1, Wemb2, bemb2, dst, xf32);
  f32_to_f16_kernel<<<featBlocks, 256, 0, stream>>>(xf32, xf16, nfeat);

  for (int l = 0; l < 3; ++l) {
    zero_kernel<<<featBlocks, 256, 0, stream>>>(agg, nfeat);
    conv_edge_kernel<<<edgeBlocks, 256, 0, stream>>>(
        d, fW1 + l * HDIM * HDIM, fB1 + l * HDIM,
        fW2 + l * HDIM * HDIM, fB2 + l * HDIM,
        Win + l * HDIM * HDIM, Bin + l * HDIM,
        src, dst, xf16, agg);
    conv_node_kernel<<<nodeBlocks, 256, 0, stream>>>(
        agg, Wout + l * HDIM * HDIM, Bout + l * HDIM, xf16, xf32);
  }

  pool_kernel<<<featBlocks, 256, 0, stream>>>(xf32, gid, pooled, counts);
  head_kernel<<<1, 64, 0, stream>>>(pooled, counts, Wfc, bfc, (float*)d_out);
}
